// Attention_31164282700397
// MI455X (gfx1250) — matrix operand order repaired and checked by reading, not compiled, not run on hardware
//
#include <hip/hip_runtime.h>
#include <hip/hip_bf16.h>

// ---------------- problem constants ----------------
#define EMBED   768
#define NQKV    2304
#define NSEQ    1024
#define BATCH   8
#define HEADS   12
#define HDIM    64
#define MROWS   (BATCH * NSEQ)          // 8192

typedef __attribute__((ext_vector_type(16))) __bf16 bf16x16;
typedef __attribute__((ext_vector_type(8)))  float  f32x8;
typedef __attribute__((ext_vector_type(4)))  unsigned int u32x4;
typedef __attribute__((ext_vector_type(8)))  int  i32x8;
typedef __attribute__((ext_vector_type(4)))  int  i32x4;

union FragB16 { bf16x16 v; unsigned u[8]; };
union FragF32 { f32x8  v; float f[8]; };

__device__ __forceinline__ unsigned short f2bf(float x) {
    unsigned u = __float_as_uint(x);
    unsigned r = (u + 0x7FFFu + ((u >> 16) & 1u)) >> 16;   // RNE
    return (unsigned short)r;
}
__device__ __forceinline__ unsigned pack_bf16(float lo, float hi) {
    return (unsigned)f2bf(lo) | ((unsigned)f2bf(hi) << 16);
}

// A-fragment K offset for 16x32 bf16 A matrix (ISA 7.12.2)
__device__ __forceinline__ int a_koff(int v, int hi) {
    return ((v < 4) ? 0 : 16) + hi * 8 + (v & 3) * 2;
}

// --- TDM: issue a 2-D tensor_load_to_lds (D# per ISA 08_async_tensor §8) ---
// dims/strides in data_size (=2B) units; lds_off/global addr in bytes.
// pad_interval: LDS padding every 2^(code+1) DWORDs; pad_amount: (code+1) DWORDs.
__device__ __forceinline__ void tdm_load_2d(unsigned lds_off, unsigned long long ga,
                                            unsigned dim0, unsigned dim1,
                                            unsigned stride0,
                                            unsigned tile0, unsigned tile1,
                                            unsigned pad_interval, unsigned pad_amount)
{
    u32x4 g0;
    g0.x = 1u;                                              // count=1, user mode
    g0.y = lds_off;                                         // lds_addr
    g0.z = (unsigned)ga;                                    // global_addr[31:0]
    g0.w = (unsigned)((ga >> 32) & 0x1FFFFFFu) | 0x80000000u; // ga[56:32] | type=2
    i32x8 g1;
    g1[0] = (int)((1u << 16)                 // data_size = 2 bytes
                | (1u << 20)                 // pad_enable
                | (pad_interval << 22)
                | (pad_amount << 25));
    g1[1] = (int)(dim0 << 16);                              // tensor_dim0[15:0]
    g1[2] = (int)((dim0 >> 16) | (dim1 << 16));             // dim0 hi | dim1 lo
    g1[3] = (int)((dim1 >> 16) | (tile0 << 16));            // dim1 hi | tile_dim0
    g1[4] = (int)(tile1 & 0xFFFFu);                         // tile_dim1 (tile_dim2=0)
    g1[5] = (int)stride0;                                   // tensor_dim0_stride lo
    g1[6] = 0;
    g1[7] = 0;
    i32x4 gz = {0, 0, 0, 0};                                // dims >=2 unused
    asm volatile("tensor_load_to_lds %0, %1, %2, %3"
                 :: "s"(g0), "s"(g1), "s"(gz), "s"(gz)
                 : "memory");
}

__device__ __forceinline__ void wait_tensorcnt_0() {
#if __has_builtin(__builtin_amdgcn_s_wait_tensorcnt)
    __builtin_amdgcn_s_wait_tensorcnt(0);
#else
    asm volatile("s_wait_tensorcnt 0x0" ::: "memory");
#endif
}
__device__ __forceinline__ void wait_tensorcnt_2() {
#if __has_builtin(__builtin_amdgcn_s_wait_tensorcnt)
    __builtin_amdgcn_s_wait_tensorcnt(2);
#else
    asm volatile("s_wait_tensorcnt 0x2" ::: "memory");
#endif
}

// =====================================================================
// Kernel 1: qkv = x @ W_qkv + b_qkv, scattered to q/k [B,H,N,D] bf16 and
//           v transposed [B,H,D,N] bf16.
// =====================================================================
__global__ __launch_bounds__(256)
void qkv_gemm_kernel(const float* __restrict__ x,
                     const float* __restrict__ Wqkv,
                     const float* __restrict__ bqkv,
                     __bf16* __restrict__ qb,
                     __bf16* __restrict__ kbuf,
                     __bf16* __restrict__ vbuf)
{
    constexpr int BM = 128, BN = 64, BK = 32;
    constexpr int AS = 40;   // As row stride (halves), 80B: 16B aligned
    constexpr int BS = 34;   // Bs row stride (halves), Bs[n][k] packed
    __shared__ __bf16 As[BM * AS];
    __shared__ __bf16 Bs[BN * BS];

    const int t    = threadIdx.x;
    const int lane = t & 31;
    const int wave = t >> 5;
    const int m    = lane & 15;
    const int hi   = lane >> 4;
    const int nb   = blockIdx.x * BN;
    const int mb   = blockIdx.y * BM;

    FragF32 acc[4];
#pragma unroll
    for (int ct = 0; ct < 4; ct++)
#pragma unroll
        for (int j = 0; j < 8; j++) acc[ct].f[j] = 0.0f;

    const int arow  = t >> 1;          // 0..127
    const int ahalf = (t & 1) * 16;    // 0 or 16 (cols)
    const int brow  = t >> 3;          // k 0..31
    const int bcol  = (t & 7) * 8;     // col group

    for (int kb0 = 0; kb0 < EMBED; kb0 += BK) {
        __syncthreads();
        {   // A tile: 128 x 32 fp32 -> bf16 LDS
            const float4* src = (const float4*)(x + (size_t)(mb + arow) * EMBED + kb0 + ahalf);
            uint2* dst = (uint2*)(As + arow * AS + ahalf);
#pragma unroll
            for (int j = 0; j < 4; j++) {
                float4 f = src[j];
                dst[j] = make_uint2(pack_bf16(f.x, f.y), pack_bf16(f.z, f.w));
            }
        }
        {   // B tile: 32 x 64 fp32, stored column-major-packed Bs[n][k]
            const float* src = Wqkv + (size_t)(kb0 + brow) * NQKV + nb + bcol;
            float4 f0 = *(const float4*)(src);
            float4 f1 = *(const float4*)(src + 4);
            float vals[8] = { f0.x, f0.y, f0.z, f0.w, f1.x, f1.y, f1.z, f1.w };
#pragma unroll
            for (int j = 0; j < 8; j++)
                Bs[(bcol + j) * BS + brow] = (__bf16)vals[j];
        }
        __syncthreads();

        FragB16 af;
#pragma unroll
        for (int v = 0; v < 8; v++)
            af.u[v] = *(const unsigned*)(As + (wave * 16 + m) * AS + a_koff(v, hi));

#pragma unroll
        for (int ct = 0; ct < 4; ct++) {
            FragB16 bf;
            const int n = ct * 16 + m;
#pragma unroll
            for (int v = 0; v < 8; v++)
                bf.u[v] = *(const unsigned*)(Bs + n * BS + a_koff(v, hi));
            acc[ct].v = __builtin_amdgcn_wmma_f32_16x16x32_bf16(
                false, af.v, false, bf.v, (short)0, acc[ct].v, false, false);
        }
    }

    const int which = nb / EMBED;          // 0=q 1=k 2=v (uniform per block)
    const int h     = (nb % EMBED) / HDIM; // uniform per block
#pragma unroll
    for (int ct = 0; ct < 4; ct++) {
        const int c    = nb + ct * 16 + m;
        const int d    = ct * 16 + m;
        const float bias = bqkv[c];
#pragma unroll
        for (int v = 0; v < 8; v++) {
            const int row  = mb + wave * 16 + v + 8 * hi;
            const int b    = row >> 10;
            const int nseq = row & (NSEQ - 1);
            const int bh   = b * HEADS + h;
            const __bf16 val = (__bf16)(acc[ct].f[v] + bias);
            if (which == 0)      qb  [((size_t)bh * NSEQ + nseq) * HDIM + d] = val;
            else if (which == 1) kbuf[((size_t)bh * NSEQ + nseq) * HDIM + d] = val;
            else                 vbuf[((size_t)bh * HDIM + d) * NSEQ + nseq] = val;
        }
    }
}

// =====================================================================
// Kernel 2: flash attention per (b,h) x 128-row slab.
// K/V streamed to LDS by the Tensor Data Mover, double-buffered so the
// DMA of chunk ch+1 overlaps WMMA compute of chunk ch (TENSORcnt sync).
// =====================================================================
#define ATT_CHUNK 256
#define CHUNKS    (NSEQ / ATT_CHUNK)
#define KSTR 72     // K rows stride (halves) = 64 + TDM pad(4 dw / 32 dw)
#define VSTR 264    // V rows stride (halves) = 256 + TDM pad(4 dw / 128 dw)
#define PSTR 34
#define KH (ATT_CHUNK * KSTR)   // halves per K buffer
#define VH (HDIM * VSTR)        // halves per V buffer
#define BUFH (KH + VH)
#define ATT_LDS_BYTES ((2 * BUFH + 8 * 16 * PSTR) * 2)

__global__ __launch_bounds__(256)
void attn_kernel(const __bf16* __restrict__ qb,
                 const __bf16* __restrict__ kbuf,
                 const __bf16* __restrict__ vbuf,
                 const float*  __restrict__ pos,
                 __bf16* __restrict__ obuf)
{
    extern __shared__ __bf16 smem[];
    __bf16* Ps = smem + 2 * BUFH;                   // [8 waves][16][PSTR]

    const int t    = threadIdx.x;
    const int lane = t & 31;
    const int wave = t >> 5;
    const int m    = lane & 15;
    const int hi   = lane >> 4;
    const int bh   = blockIdx.x;          // 0..95
    const int b    = bh / HEADS;
    const int h    = bh % HEADS;
    const int rowbase = blockIdx.y * 128 + wave * 16;

    // Q fragments (16 rows x 64 dims -> 2 A-frags)
    FragB16 qf[2];
    {
        const __bf16* qrow = qb + ((size_t)bh * NSEQ + rowbase + m) * HDIM;
#pragma unroll
        for (int j = 0; j < 2; j++)
#pragma unroll
            for (int v = 0; v < 8; v++)
                qf[j].u[v] = *(const unsigned*)(qrow + j * 32 + a_koff(v, hi));
    }

    float mrun[8], lrun[8];
    FragF32 oacc[4];
#pragma unroll
    for (int v = 0; v < 8; v++) { mrun[v] = -3.0e38f; lrun[v] = 0.0f; }
#pragma unroll
    for (int ct = 0; ct < 4; ct++)
#pragma unroll
        for (int j = 0; j < 8; j++) oacc[ct].f[j] = 0.0f;

    __bf16* Pw = Ps + wave * 16 * PSTR;
    const float* posrow0 = pos + ((size_t)h * NSEQ + rowbase) * NSEQ;

    // TDM issue for chunk ch into LDS buffer ch&1 (wave 0 only; uniform branch)
    auto issue_chunk = [&](int ch) {
        const int bsel = ch & 1;
        const unsigned koff = (unsigned)(unsigned long long)(size_t)(smem + bsel * BUFH);
        const unsigned voff = (unsigned)(unsigned long long)(size_t)(smem + bsel * BUFH + KH);
        const unsigned long long gaK =
            (unsigned long long)(size_t)(kbuf + ((size_t)bh * NSEQ + ch * ATT_CHUNK) * HDIM);
        const unsigned long long gaV =
            (unsigned long long)(size_t)(vbuf + (size_t)bh * HDIM * NSEQ + ch * ATT_CHUNK);
        // K: 256 rows x 64 halves, row stride 64 -> padded to 72 in LDS
        tdm_load_2d(koff, gaK, HDIM, NSEQ, HDIM, HDIM, ATT_CHUNK, /*pi*/4, /*pa*/3);
        // V: 64 rows x 256-half slice of 1024-wide rows -> padded to 264 in LDS
        tdm_load_2d(voff, gaV, NSEQ, HDIM, NSEQ, ATT_CHUNK, HDIM, /*pi*/6, /*pa*/3);
    };

    if (wave == 0) issue_chunk(0);

    for (int ch = 0; ch < CHUNKS; ch++) {
        __syncthreads();   // all waves done with buffer (ch+1)&1 from chunk ch-1
        if (wave == 0) {
            if (ch + 1 < CHUNKS) { issue_chunk(ch + 1); wait_tensorcnt_2(); }
            else                 { wait_tensorcnt_0(); }
        }
        __syncthreads();   // buffer ch&1 is now valid for everyone
        const __bf16* Ks = smem + (ch & 1) * BUFH;
        const __bf16* Vs = Ks + KH;

        for (int kb0 = 0; kb0 < ATT_CHUNK; kb0 += 32) {
            // ---- scores: two 16x16 tiles (keys kb0..+15, +16..+31) ----
            FragF32 s0, s1;
#pragma unroll
            for (int j = 0; j < 8; j++) { s0.f[j] = 0.0f; s1.f[j] = 0.0f; }
#pragma unroll
            for (int half = 0; half < 2; half++) {
                FragF32* sp = half ? &s1 : &s0;
#pragma unroll
                for (int j = 0; j < 2; j++) {   // K dim (d) 0..31, 32..63
                    FragB16 bk;
#pragma unroll
                    for (int v = 0; v < 8; v++)
                        bk.u[v] = *(const unsigned*)(Ks + (kb0 + half * 16 + m) * KSTR
                                                     + j * 32 + a_koff(v, hi));
                    sp->v = __builtin_amdgcn_wmma_f32_16x16x32_bf16(
                        false, qf[j].v, false, bk.v, (short)0, sp->v, false, false);
                }
            }
            // ---- bias (pre-scale, as in reference) + scale ----
            const int keyb = ch * ATT_CHUNK + kb0;
#pragma unroll
            for (int v = 0; v < 8; v++) {
                const float* pr = posrow0 + (size_t)(v + 8 * hi) * NSEQ + keyb + m;
                s0.f[v] = (s0.f[v] + pr[0])  * 0.125f;
                s1.f[v] = (s1.f[v] + pr[16]) * 0.125f;
            }
            // ---- online softmax update ----
            float tm[8];
#pragma unroll
            for (int v = 0; v < 8; v++) tm[v] = fmaxf(s0.f[v], s1.f[v]);
#pragma unroll
            for (int off = 1; off < 16; off <<= 1)
#pragma unroll
                for (int v = 0; v < 8; v++)
                    tm[v] = fmaxf(tm[v], __shfl_xor(tm[v], off, 32));
            float corr[8];
#pragma unroll
            for (int v = 0; v < 8; v++) {
                float nm = fmaxf(mrun[v], tm[v]);
                corr[v]  = __expf(mrun[v] - nm);
                mrun[v]  = nm;
                s0.f[v]  = __expf(s0.f[v] - nm);
                s1.f[v]  = __expf(s1.f[v] - nm);
            }
            float ts[8];
#pragma unroll
            for (int v = 0; v < 8; v++) ts[v] = s0.f[v] + s1.f[v];
#pragma unroll
            for (int off = 1; off < 16; off <<= 1)
#pragma unroll
                for (int v = 0; v < 8; v++) ts[v] += __shfl_xor(ts[v], off, 32);
#pragma unroll
            for (int v = 0; v < 8; v++) lrun[v] = lrun[v] * corr[v] + ts[v];
#pragma unroll
            for (int ct = 0; ct < 4; ct++)
#pragma unroll
                for (int v = 0; v < 8; v++) oacc[ct].f[v] *= corr[v];

            // ---- stage P (C-layout -> A-layout via per-wave LDS) ----
#pragma unroll
            for (int v = 0; v < 8; v++) {
                const int r = v + 8 * hi;
                Pw[r * PSTR + m]      = (__bf16)s0.f[v];
                Pw[r * PSTR + 16 + m] = (__bf16)s1.f[v];
            }
            FragB16 pf;
#pragma unroll
            for (int v = 0; v < 8; v++)
                pf.u[v] = *(const unsigned*)(Pw + m * PSTR + a_koff(v, hi));

            // ---- O += P @ V ----
#pragma unroll
            for (int ct = 0; ct < 4; ct++) {
                FragB16 bv;
#pragma unroll
                for (int v = 0; v < 8; v++)
                    bv.u[v] = *(const unsigned*)(Vs + (ct * 16 + m) * VSTR
                                                 + kb0 + a_koff(v, hi));
                oacc[ct].v = __builtin_amdgcn_wmma_f32_16x16x32_bf16(
                    false, pf.v, false, bv.v, (short)0, oacc[ct].v, false, false);
            }
        }
    }

    // epilogue: normalize, write bf16 [B,N,H*D]
    float inv[8];
#pragma unroll
    for (int v = 0; v < 8; v++) inv[v] = 1.0f / lrun[v];
#pragma unroll
    for (int ct = 0; ct < 4; ct++)
#pragma unroll
        for (int v = 0; v < 8; v++) {
            const int row = rowbase + v + 8 * hi;
            obuf[((size_t)b * NSEQ + row) * EMBED + h * HDIM + ct * 16 + m] =
                (__bf16)(oacc[ct].f[v] * inv[v]);
        }
}

// =====================================================================
// Kernel 3: out = attn_out(bf16) @ W_proj + b_proj  (fp32 out)
// =====================================================================
__global__ __launch_bounds__(256)
void proj_gemm_kernel(const __bf16* __restrict__ a,
                      const float*  __restrict__ Wproj,
                      const float*  __restrict__ bproj,
                      float* __restrict__ out)
{
    constexpr int BM = 128, BN = 64, BK = 32;
    constexpr int AS = 40, BS = 34;
    __shared__ __bf16 As[BM * AS];
    __shared__ __bf16 Bs[BN * BS];

    const int t    = threadIdx.x;
    const int lane = t & 31;
    const int wave = t >> 5;
    const int m    = lane & 15;
    const int hi   = lane >> 4;
    const int nb   = blockIdx.x * BN;
    const int mb   = blockIdx.y * BM;

    FragF32 acc[4];
#pragma unroll
    for (int ct = 0; ct < 4; ct++)
#pragma unroll
        for (int j = 0; j < 8; j++) acc[ct].f[j] = 0.0f;

    const int arow  = t >> 1;
    const int ahalf = (t & 1) * 16;
    const int brow  = t >> 3;
    const int bcol  = (t & 7) * 8;

    for (int kb0 = 0; kb0 < EMBED; kb0 += BK) {
        __syncthreads();
        {   // A already bf16: straight 2 x b128 copy per thread
            const uint4* src = (const uint4*)(a + (size_t)(mb + arow) * EMBED + kb0 + ahalf);
            uint4* dst = (uint4*)(As + arow * AS + ahalf);
            dst[0] = src[0];
            dst[1] = src[1];
        }
        {
            const float* src = Wproj + (size_t)(kb0 + brow) * EMBED + nb + bcol;
            float4 f0 = *(const float4*)(src);
            float4 f1 = *(const float4*)(src + 4);
            float vals[8] = { f0.x, f0.y, f0.z, f0.w, f1.x, f1.y, f1.z, f1.w };
#pragma unroll
            for (int j = 0; j < 8; j++)
                Bs[(bcol + j) * BS + brow] = (__bf16)vals[j];
        }
        __syncthreads();

        FragB16 af;
#pragma unroll
        for (int v = 0; v < 8; v++)
            af.u[v] = *(const unsigned*)(As + (wave * 16 + m) * AS + a_koff(v, hi));
#pragma unroll
        for (int ct = 0; ct < 4; ct++) {
            FragB16 bf;
            const int n = ct * 16 + m;
#pragma unroll
            for (int v = 0; v < 8; v++)
                bf.u[v] = *(const unsigned*)(Bs + n * BS + a_koff(v, hi));
            acc[ct].v = __builtin_amdgcn_wmma_f32_16x16x32_bf16(
                false, af.v, false, bf.v, (short)0, acc[ct].v, false, false);
        }
    }

#pragma unroll
    for (int ct = 0; ct < 4; ct++) {
        const int c = nb + ct * 16 + m;
        const float bias = bproj[c];
#pragma unroll
        for (int v = 0; v < 8; v++) {
            const int row = mb + wave * 16 + v + 8 * hi;
            out[(size_t)row * EMBED + c] = acc[ct].f[v] + bias;
        }
    }
}

// =====================================================================
extern "C" void kernel_launch(void* const* d_in, const int* in_sizes, int n_in,
                              void* d_out, int out_size, void* d_ws, size_t ws_size,
                              hipStream_t stream) {
    const float* x     = (const float*)d_in[0];
    const float* pos   = (const float*)d_in[1];
    const float* Wqkv  = (const float*)d_in[2];
    const float* bqkv  = (const float*)d_in[3];
    const float* Wproj = (const float*)d_in[4];
    const float* bproj = (const float*)d_in[5];
    float* out = (float*)d_out;

    const size_t PLANE = (size_t)BATCH * HEADS * NSEQ * HDIM;  // 6,291,456 halves
    __bf16* qb   = (__bf16*)d_ws;
    __bf16* kbuf = qb + PLANE;
    __bf16* vbuf = qb + 2 * PLANE;
    __bf16* abuf = qb + 3 * PLANE;

    // allow >64KB dynamic LDS for the attention kernel (host-side, capture-safe)
    (void)hipFuncSetAttribute((const void*)attn_kernel,
                              hipFuncAttributeMaxDynamicSharedMemorySize,
                              ATT_LDS_BYTES);

    qkv_gemm_kernel<<<dim3(NQKV / 64, MROWS / 128), 256, 0, stream>>>(
        x, Wqkv, bqkv, qb, kbuf, vbuf);

    attn_kernel<<<dim3(BATCH * HEADS, NSEQ / 128), 256, ATT_LDS_BYTES, stream>>>(
        qb, kbuf, vbuf, pos, abuf);

    proj_gemm_kernel<<<dim3(EMBED / 64, MROWS / 128), 256, 0, stream>>>(
        abuf, Wproj, bproj, out);
}
